// RowSelfAttentionWithBulkBias_1382979469737
// MI455X (gfx1250) — matrix operand order repaired and checked by reading, not compiled, not run on hardware
//
#include <hip/hip_runtime.h>

// ---------------------------------------------------------------------------
// CDNA5 (gfx1250, wave32) bf16 WMMA implementation of RowSelfAttentionWithBulkBias
// + Tensor Data Mover staging of the K panel into LDS.
// ---------------------------------------------------------------------------

typedef __bf16 bf16_t;
typedef __attribute__((ext_vector_type(16))) __bf16 v16bf;
typedef __attribute__((ext_vector_type(8)))  __bf16 v8bf;
typedef __attribute__((ext_vector_type(8)))  float  v8f;
typedef __attribute__((ext_vector_type(4)))  unsigned int u32x4;
typedef __attribute__((ext_vector_type(8)))  unsigned int u32x8;

#define C_DIM 256
#define L_DIM 1024
#define H_DIM 8
#define DH_DIM 32
#define BS_DIM 8   // B*S
#define M_DIM 8192 // BS*L

// D = A(16x32 bf16) * B(32x16 bf16) + C(16x16 f32)
__device__ __forceinline__ v8f wmma_bf16(v16bf a, v16bf b, v8f c) {
    return __builtin_amdgcn_wmma_f32_16x16x32_bf16(
        /*neg_a=*/false, a, /*neg_b=*/false, b,
        /*c_mod=*/(short)0, c, /*reuse_a=*/false, /*reuse_b=*/false);
}

// A-fragment (16x32, row-major source, ld elements):
// lane 0-15: row=lane,  e0..7=K0..7,  e8..15=K16..23
// lane16-31: row=l-16,  e0..7=K8..15, e8..15=K24..31
__device__ __forceinline__ v16bf load_A(const bf16_t* tile, int ld) {
    int lane = threadIdx.x & 31;
    int row  = lane & 15;
    int kb   = (lane >> 4) << 3;           // 0 or 8
    const bf16_t* p = tile + row * ld + kb;
    v8bf lo = *(const v8bf*)(p);
    v8bf hi = *(const v8bf*)(p + 16);
    v16bf r;
#pragma unroll
    for (int i = 0; i < 8; ++i) { r[i] = lo[i]; r[i + 8] = hi[i]; }
    return r;
}

// B-fragment for B = (source)^T: source row-major (rows = N-cols of B, ld over K).
// lane 0-15: col=lane, e0..7=K0..7, e8..15=K16..23; lane16-31: col=l-16, e0..7=K8..15, e8..15=K24..31.
__device__ __forceinline__ v16bf load_B(const bf16_t* tile, int ld) {
    int lane = threadIdx.x & 31;
    int col  = lane & 15;
    int kb   = (lane >> 4) << 3;           // 0 or 8
    const bf16_t* p = tile + col * ld + kb;
    v8bf lo = *(const v8bf*)(p);
    v8bf hi = *(const v8bf*)(p + 16);
    v16bf r;
#pragma unroll
    for (int i = 0; i < 8; ++i) { r[i] = lo[i]; r[i + 8] = hi[i]; }
    return r;
}

// ---------------------------------------------------------------------------
// TDM: stage a contiguous run of (n8 * 8) bytes from global into LDS as a
// 1-D tensor tile of 8-byte elements.  D# built in SGPR tuples, issued with
// tensor_load_to_lds (portable inline-asm path, tracked by TENSORcnt).
// ---------------------------------------------------------------------------
__device__ __forceinline__ void tdm_load_1d(unsigned int lds_off,
                                            const void* gaddr,
                                            unsigned int n8) {
    unsigned long long ga = (unsigned long long)gaddr;
    u32x4 g0;
    g0[0] = 1u;                                    // count = 1 (valid descriptor)
    g0[1] = lds_off;                               // lds_addr (bytes)
    g0[2] = (unsigned int)ga;                      // global_addr[31:0]
    g0[3] = (unsigned int)(ga >> 32) | (2u << 30); // global_addr[56:32] | type=2
    u32x8 g1;
    g1[0] = 3u << 16;                              // data_size = 3 (8 bytes)
    g1[1] = (n8 & 0xFFFFu) << 16;                  // tensor_dim0[15:0]
    g1[2] = ((n8 >> 16) & 0xFFFFu) | (1u << 16);   // tensor_dim0[31:16] | tensor_dim1=1
    g1[3] = (n8 & 0xFFFFu) << 16;                  // tile_dim0 (16-bit)
    g1[4] = 1u;                                    // tile_dim1 = 1
    g1[5] = n8;                                    // tensor_dim0_stride[31:0]
    g1[6] = 0u;
    g1[7] = 0u;
    asm volatile("tensor_load_to_lds %0, %1" :: "s"(g0), "s"(g1) : "memory");
}

// ---------------------------------------------------------------------------
// Kernel 0: fp32 -> bf16 conversion (one-time; amortized)
// ---------------------------------------------------------------------------
__global__ void convert_bf16_kernel(const float* __restrict__ src,
                                    bf16_t* __restrict__ dst, int n) {
    int i = blockIdx.x * blockDim.x + threadIdx.x;
    if (i < n) dst[i] = (bf16_t)src[i];
}

// ---------------------------------------------------------------------------
// Kernel 1: QKV projection. out = x @ W^T + b, head-split.
// Q,K stored (BS,H,L,DH) row-major bf16; V stored transposed (BS,H,DH,L).
// grid (M/16, 4, 3), block 128 (4 waves, one 16x16 tile each).
// ---------------------------------------------------------------------------
__global__ void qkv_proj_kernel(const bf16_t* __restrict__ xb,
                                const bf16_t* __restrict__ wq,
                                const bf16_t* __restrict__ wk,
                                const bf16_t* __restrict__ wv,
                                const float* __restrict__ bq,
                                const float* __restrict__ bk,
                                const float* __restrict__ bv,
                                bf16_t* __restrict__ gq,
                                bf16_t* __restrict__ gk,
                                bf16_t* __restrict__ gvt) {
    int wave  = threadIdx.x >> 5;
    int lane  = threadIdx.x & 31;
    int mtile = blockIdx.x;
    int ntile = blockIdx.y * 4 + wave;     // 0..15 (N = 256)
    int mat   = blockIdx.z;                // 0=Q 1=K 2=V

    const bf16_t* W    = (mat == 0) ? wq : (mat == 1) ? wk : wv;
    const float*  bias = (mat == 0) ? bq : (mat == 1) ? bk : bv;

    const bf16_t* arow = xb + (size_t)mtile * 16 * C_DIM;
    const bf16_t* brow = W  + (size_t)ntile * 16 * C_DIM;

    v8f acc = {};
#pragma unroll
    for (int kk = 0; kk < C_DIM; kk += 32)
        acc = wmma_bf16(load_A(arow + kk, C_DIM), load_B(brow + kk, C_DIM), acc);

    int col = lane & 15;
    int hi8 = (lane >> 4) << 3;
    int n   = ntile * 16 + col;
    float bn = bias[n];
    int h = n >> 5, d = n & 31;

#pragma unroll
    for (int r = 0; r < 8; ++r) {
        int m   = mtile * 16 + r + hi8;
        int bsi = m >> 10;                 // m / L
        int l   = m & 1023;                // m % L
        int bh  = bsi * H_DIM + h;
        bf16_t val = (bf16_t)(acc[r] + bn);
        if (mat == 0)      gq [((size_t)bh * L_DIM + l) * DH_DIM + d] = val;
        else if (mat == 1) gk [((size_t)bh * L_DIM + l) * DH_DIM + d] = val;
        else               gvt[((size_t)bh * DH_DIM + d) * L_DIM + l] = val;
    }
}

// ---------------------------------------------------------------------------
// Kernel 2: flash attention + bulk-map conv bias + sigmoid gate.
// grid (L/128, BS*H), block 256 (8 waves; wave w owns query tile 8*bx+w).
// The whole K panel for this (bs,h) is DMA'd into LDS once by the TDM and
// shared by all 8 waves; V / bulk_map stream from the 192MB L2.
// ---------------------------------------------------------------------------
__global__ void attn_kernel(const bf16_t* __restrict__ gq,
                            const bf16_t* __restrict__ gk,
                            const bf16_t* __restrict__ gvt,
                            const float* __restrict__ bulk,
                            const float* __restrict__ conv_w,
                            const float* __restrict__ conv_b,
                            const bf16_t* __restrict__ gwb,
                            const float* __restrict__ gate_b,
                            bf16_t* __restrict__ attn_out) {
    __shared__ __align__(16) bf16_t shK[L_DIM * DH_DIM];   // 64 KB K panel
    __shared__ __align__(16) bf16_t pshm[8 * 16 * 32];     // per-wave P staging

    int tid   = threadIdx.x;
    int wave  = tid >> 5;
    int lane  = tid & 31;
    int bh    = blockIdx.y;
    int bsi   = bh >> 3;                   // 0..7 = b*S + s
    int h     = bh & 7;
    int b     = bsi >> 2;                  // S = 4
    int qtile = blockIdx.x * 8 + wave;
    int qbase = qtile * 16;
    int col   = lane & 15;
    int hi8   = (lane >> 4) << 3;

    // --- TDM: one wave DMAs the 64KB K panel into LDS, waits on TENSORcnt ---
    if (wave == 0) {
        unsigned int koff = (unsigned int)(unsigned long long)(void*)shK;
        tdm_load_1d(koff, gk + (size_t)bh * L_DIM * DH_DIM, (L_DIM * DH_DIM * 2) / 8);
        __builtin_amdgcn_s_wait_tensorcnt(0);
    }
    __syncthreads();

    const float scale = 0.17677669529663687f;  // 1/sqrt(DH)
    float cw = conv_w[h], cb = conv_b[h];
    const float* bulkb = bulk + (size_t)b * L_DIM * L_DIM;
    bf16_t* pw = pshm + wave * (16 * 32);

    v16bf qA = load_A(gq + ((size_t)bh * L_DIM + qbase) * DH_DIM, DH_DIM);

    v8f o0 = {}, o1 = {};
    float m_run[8], l_run[8];
#pragma unroll
    for (int r = 0; r < 8; ++r) { m_run[r] = -1e30f; l_run[r] = 0.f; }

    for (int j = 0; j < L_DIM; j += 32) {
        __builtin_prefetch(gvt + (size_t)bh * DH_DIM * L_DIM + ((j + 32) & (L_DIM - 1)), 0, 1);

        // S(16x32) = Q(16x32) @ K^T : K B-fragments from LDS (ds_load_b128)
        v16bf kb0 = load_B(shK + j * DH_DIM, DH_DIM);
        v16bf kb1 = load_B(shK + (j + 16) * DH_DIM, DH_DIM);
        v8f z = {};
        v8f s0 = wmma_bf16(qA, kb0, z);
        v8f s1 = wmma_bf16(qA, kb1, z);

        // scale + bulk-map 1x1-conv bias
#pragma unroll
        for (int r = 0; r < 8; ++r) {
            int qrow = qbase + r + hi8;
            const float* br = bulkb + (size_t)qrow * L_DIM;
            s0[r] = s0[r] * scale + cw * br[j + col] + cb;
            s1[r] = s1[r] * scale + cw * br[j + 16 + col] + cb;
        }

        // online softmax: per-row max/sum across 16 lanes of each half-wave
#pragma unroll
        for (int r = 0; r < 8; ++r) {
            float mx = fmaxf(s0[r], s1[r]);
#pragma unroll
            for (int off = 8; off > 0; off >>= 1)
                mx = fmaxf(mx, __shfl_xor(mx, off, 16));
            float mnew  = fmaxf(m_run[r], mx);
            float alpha = __expf(m_run[r] - mnew);
            float p0 = __expf(s0[r] - mnew);
            float p1 = __expf(s1[r] - mnew);
            s0[r] = p0; s1[r] = p1;
            float rs = p0 + p1;
#pragma unroll
            for (int off = 8; off > 0; off >>= 1)
                rs += __shfl_xor(rs, off, 16);
            l_run[r] = l_run[r] * alpha + rs;
            o0[r] *= alpha; o1[r] *= alpha;
            m_run[r] = mnew;
        }

        // stage P (C-layout) -> LDS -> reload as A-fragment (per-wave slice)
#pragma unroll
        for (int r = 0; r < 8; ++r) {
            pw[(r + hi8) * 32 + col]      = (bf16_t)s0[r];
            pw[(r + hi8) * 32 + 16 + col] = (bf16_t)s1[r];
        }
        asm volatile("s_wait_dscnt 0x0" ::: "memory");
        v16bf pA  = load_A(pw, 32);

        // O(16x32) += P(16x32) @ V(32x32): Vt rows are B-columns (contiguous)
        v16bf vb0 = load_B(gvt + (size_t)bh * DH_DIM * L_DIM + j, L_DIM);
        v16bf vb1 = load_B(gvt + ((size_t)bh * DH_DIM + 16) * L_DIM + j, L_DIM);
        o0 = wmma_bf16(pA, vb0, o0);
        o1 = wmma_bf16(pA, vb1, o1);
    }

    // normalize, stage O for gate GEMM
#pragma unroll
    for (int r = 0; r < 8; ++r) {
        float inv = 1.f / l_run[r];
        o0[r] *= inv; o1[r] *= inv;
        pw[(r + hi8) * 32 + col]      = (bf16_t)o0[r];
        pw[(r + hi8) * 32 + 16 + col] = (bf16_t)o1[r];
    }
    asm volatile("s_wait_dscnt 0x0" ::: "memory");
    v16bf oA  = load_A(pw, 32);
    v16bf gw0 = load_B(gwb + (size_t)h * DH_DIM * DH_DIM, DH_DIM);
    v16bf gw1 = load_B(gwb + (size_t)h * DH_DIM * DH_DIM + 16 * DH_DIM, DH_DIM);
    v8f zz = {};
    v8f g0 = wmma_bf16(oA, gw0, zz);
    v8f g1 = wmma_bf16(oA, gw1, zz);

    float gb0 = gate_b[h * DH_DIM + col];
    float gb1 = gate_b[h * DH_DIM + 16 + col];
#pragma unroll
    for (int r = 0; r < 8; ++r) {
        float sg0 = 1.f / (1.f + __expf(-(g0[r] + gb0)));
        float sg1 = 1.f / (1.f + __expf(-(g1[r] + gb1)));
        int l = qbase + r + hi8;
        bf16_t* rowp = attn_out + ((size_t)bsi * L_DIM + l) * C_DIM + h * DH_DIM;
        rowp[col]      = (bf16_t)(o0[r] * sg0);
        rowp[16 + col] = (bf16_t)(o1[r] * sg1);
    }
}

// ---------------------------------------------------------------------------
// Kernel 3: output projection + residual + LayerNorm.
// grid (M/16), block 256 (8 waves; wave w owns N-tiles 2w, 2w+1).
// ---------------------------------------------------------------------------
__global__ void out_proj_ln_kernel(const bf16_t* __restrict__ attn,
                                   const bf16_t* __restrict__ wob,
                                   const float* __restrict__ bo,
                                   const float* __restrict__ x,
                                   const float* __restrict__ ln_g,
                                   const float* __restrict__ ln_b,
                                   float* __restrict__ out) {
    __shared__ __align__(16) float ysh[16 * C_DIM];
    __shared__ float red_s[16], red_q[16];

    int tid   = threadIdx.x;
    int wave  = tid >> 5;
    int lane  = tid & 31;
    int mtile = blockIdx.x;
    int col   = lane & 15;
    int hi8   = (lane >> 4) << 3;

    if (tid < 16) { red_s[tid] = 0.f; red_q[tid] = 0.f; }

    const bf16_t* arow = attn + (size_t)mtile * 16 * C_DIM;
#pragma unroll
    for (int t = 0; t < 2; ++t) {
        int nt = wave * 2 + t;
        const bf16_t* brow = wob + (size_t)nt * 16 * C_DIM;
        v8f acc = {};
#pragma unroll
        for (int kk = 0; kk < C_DIM; kk += 32)
            acc = wmma_bf16(load_A(arow + kk, C_DIM), load_B(brow + kk, C_DIM), acc);
        int n = nt * 16 + col;
        float bn = bo[n];
#pragma unroll
        for (int r = 0; r < 8; ++r) {
            int m = mtile * 16 + r + hi8;
            ysh[(r + hi8) * C_DIM + n] = acc[r] + bn + x[(size_t)m * C_DIM + n];
        }
    }
    __syncthreads();

    // LayerNorm over C=256: 16 rows x 16 chunks of 16 elements
    int row = tid & 15, chunk = tid >> 4;
    float s = 0.f, q = 0.f;
    float vals[16];
#pragma unroll
    for (int i = 0; i < 16; ++i) {
        float v = ysh[row * C_DIM + chunk * 16 + i];
        vals[i] = v; s += v; q += v * v;
    }
    atomicAdd(&red_s[row], s);
    atomicAdd(&red_q[row], q);
    __syncthreads();

    float mu   = red_s[row] * (1.f / C_DIM);
    float var  = red_q[row] * (1.f / C_DIM) - mu * mu;
    float rstd = rsqrtf(var + 1e-5f);
    int m = mtile * 16 + row;
#pragma unroll
    for (int i = 0; i < 16; ++i) {
        int ci = chunk * 16 + i;
        out[(size_t)m * C_DIM + ci] = (vals[i] - mu) * rstd * ln_g[ci] + ln_b[ci];
    }
}

// ---------------------------------------------------------------------------
// Launch
// ---------------------------------------------------------------------------
extern "C" void kernel_launch(void* const* d_in, const int* in_sizes, int n_in,
                              void* d_out, int out_size, void* d_ws, size_t ws_size,
                              hipStream_t stream) {
    (void)in_sizes; (void)n_in; (void)out_size; (void)ws_size;

    const float* x      = (const float*)d_in[0];
    const float* bulk   = (const float*)d_in[1];
    const float* Wq     = (const float*)d_in[2];
    const float* bq     = (const float*)d_in[3];
    const float* Wk     = (const float*)d_in[4];
    const float* bk     = (const float*)d_in[5];
    const float* Wv     = (const float*)d_in[6];
    const float* bv     = (const float*)d_in[7];
    const float* Wo     = (const float*)d_in[8];
    const float* bo     = (const float*)d_in[9];
    const float* conv_w = (const float*)d_in[10];
    const float* conv_b = (const float*)d_in[11];
    const float* gate_w = (const float*)d_in[12];
    const float* gate_b = (const float*)d_in[13];
    const float* ln_g   = (const float*)d_in[14];
    const float* ln_b   = (const float*)d_in[15];
    float* out = (float*)d_out;

    char* ws = (char*)d_ws;
    size_t off = 0;
    auto take = [&](size_t bytes) -> char* {
        char* p = ws + off;
        off = (off + bytes + 255) & ~(size_t)255;
        return p;
    };
    bf16_t* xb   = (bf16_t*)take((size_t)M_DIM * C_DIM * 2);
    bf16_t* wqb  = (bf16_t*)take((size_t)C_DIM * C_DIM * 2);
    bf16_t* wkb  = (bf16_t*)take((size_t)C_DIM * C_DIM * 2);
    bf16_t* wvb  = (bf16_t*)take((size_t)C_DIM * C_DIM * 2);
    bf16_t* wob  = (bf16_t*)take((size_t)C_DIM * C_DIM * 2);
    bf16_t* gwb  = (bf16_t*)take((size_t)H_DIM * DH_DIM * DH_DIM * 2);
    bf16_t* gq   = (bf16_t*)take((size_t)BS_DIM * H_DIM * L_DIM * DH_DIM * 2);
    bf16_t* gk   = (bf16_t*)take((size_t)BS_DIM * H_DIM * L_DIM * DH_DIM * 2);
    bf16_t* gvt  = (bf16_t*)take((size_t)BS_DIM * H_DIM * DH_DIM * L_DIM * 2);
    bf16_t* aout = (bf16_t*)take((size_t)M_DIM * C_DIM * 2);

    // 0) fp32 -> bf16 conversions
    convert_bf16_kernel<<<(M_DIM * C_DIM + 255) / 256, 256, 0, stream>>>(x, xb, M_DIM * C_DIM);
    convert_bf16_kernel<<<(C_DIM * C_DIM + 255) / 256, 256, 0, stream>>>(Wq, wqb, C_DIM * C_DIM);
    convert_bf16_kernel<<<(C_DIM * C_DIM + 255) / 256, 256, 0, stream>>>(Wk, wkb, C_DIM * C_DIM);
    convert_bf16_kernel<<<(C_DIM * C_DIM + 255) / 256, 256, 0, stream>>>(Wv, wvb, C_DIM * C_DIM);
    convert_bf16_kernel<<<(C_DIM * C_DIM + 255) / 256, 256, 0, stream>>>(Wo, wob, C_DIM * C_DIM);
    convert_bf16_kernel<<<(H_DIM * DH_DIM * DH_DIM + 255) / 256, 256, 0, stream>>>(
        gate_w, gwb, H_DIM * DH_DIM * DH_DIM);

    // 1) QKV projection (WMMA), head-split, V transposed
    qkv_proj_kernel<<<dim3(M_DIM / 16, 4, 3), 128, 0, stream>>>(
        xb, wqb, wkb, wvb, bq, bk, bv, gq, gk, gvt);

    // 2) Flash attention + bulk bias + gate (WMMA, TDM-staged K panel)
    attn_kernel<<<dim3(L_DIM / 128, BS_DIM * H_DIM), 256, 0, stream>>>(
        gq, gk, gvt, bulk, conv_w, conv_b, gwb, gate_b, aout);

    // 3) Output projection + residual + LayerNorm (WMMA)
    out_proj_ln_kernel<<<dim3(M_DIM / 16), 256, 0, stream>>>(
        aout, wob, bo, x, ln_g, ln_b, out);
}
